// WordRNN_1151051235822
// MI455X (gfx1250) — compile-verified, matrix operand order checked by reading, not run on hardware
//
#include <hip/hip_runtime.h>
#include <hip/hip_bf16.h>
#include <stdint.h>

// ---------------- problem constants (match reference) ----------------
constexpr int E    = 300;
constexpr int H    = 512;
constexpr int B    = 256;
constexpr int T    = 200;
constexpr int Epad = 320;     // E padded up to a multiple of 32 (bf16 WMMA K step)
constexpr int G3H  = 3 * H;   // 1536
constexpr int H2   = 2 * H;   // 1024
constexpr int NBLK = 64;      // persistent grid: 2 dirs x 4 M-blocks x 8 N-blocks

typedef __attribute__((ext_vector_type(16))) __bf16 v16bf;
typedef __attribute__((ext_vector_type(8)))  float  v8f;

// gfx1250 async global->LDS copy via inline asm (portable across toolchains;
// VDST = LDS byte address VGPR, VADDR = 64-bit global address, per ISA 15.18.3).
static __device__ __forceinline__ void async_load_b128(const void* gsrc, void* ldst) {
  unsigned int       loff = (unsigned int)(uintptr_t)ldst;       // LDS addr (low 32 bits)
  unsigned long long ga   = (unsigned long long)(uintptr_t)gsrc; // global address
  asm volatile("global_load_async_to_lds_b128 %0, %1, off"
               :: "v"(loff), "v"(ga) : "memory");
}

static __device__ __forceinline__ void wait_async0() {
#if defined(__has_builtin) && __has_builtin(__builtin_amdgcn_s_wait_asynccnt)
  __builtin_amdgcn_s_wait_asynccnt(0);
#else
  asm volatile("s_wait_asynccnt 0x0" ::: "memory");
#endif
}

static __device__ __forceinline__ unsigned short f2bf(float f) {
  unsigned int u = __builtin_bit_cast(unsigned int, f);
  u += 0x7FFFu + ((u >> 16) & 1u);              // round-to-nearest-even
  return (unsigned short)(u >> 16);
}

// Load one bf16 WMMA fragment from a row-major [row][k] array (global or LDS).
// Lane holds row (m or n) = rl, K chunk base = kb; 8 bf16 at kb, 8 at kb+16.
static __device__ __forceinline__ v16bf load_frag(const unsigned short* __restrict__ base,
                                                  int stride, int row, int kb) {
  union { v16bf v; float4 f[2]; } u;
  const unsigned short* p = base + (size_t)row * stride + kb;
  u.f[0] = *(const float4*)(p);
  u.f[1] = *(const float4*)(p + 16);
  return u.v;
}
static __device__ __forceinline__ v16bf load_frag_ptr(const unsigned short* p) {
  union { v16bf v; float4 f[2]; } u;
  u.f[0] = *(const float4*)(p);
  u.f[1] = *(const float4*)(p + 16);
  return u.v;
}

// ---------------- prep kernels ----------------

__global__ void cvt_pad_kernel(const float* __restrict__ src, unsigned short* __restrict__ dst,
                               int rows, int cs, int cd) {
  long i = (long)blockIdx.x * blockDim.x + threadIdx.x;
  long total = (long)rows * cd;
  if (i >= total) return;
  int r = (int)(i / cd), c = (int)(i % cd);
  dst[i] = (c < cs) ? f2bf(src[(long)r * cs + c]) : (unsigned short)0;
}

__global__ void init_state_kernel(float* __restrict__ h32, unsigned short* __restrict__ hbf,
                                  int* __restrict__ bar) {
  int i = blockIdx.x * blockDim.x + threadIdx.x;
  if (i < 2 * B * H) { h32[i] = 0.f; hbf[i] = 0; }
  if (i == 0) { bar[0] = 0; bar[1] = 0; }
}

__global__ void inv_perm_kernel(const int* __restrict__ sorted_idx, int* __restrict__ inv) {
  int p = blockIdx.x * blockDim.x + threadIdx.x;
  if (p < B) inv[sorted_idx[p]] = p;
}

// v = W_comb @ W_attn  (1024),  c = W_comb . b_attn  (scalar)
__global__ void fold_attn_kernel(const float* __restrict__ W_attn, const float* __restrict__ b_attn,
                                 const float* __restrict__ W_comb, float* __restrict__ v,
                                 float* __restrict__ c) {
  int h = blockIdx.x * blockDim.x + threadIdx.x;
  if (h < H2) {
    float s = 0.f;
    for (int j = 0; j < H2; ++j) s += W_comb[j] * W_attn[(size_t)j * H2 + h];
    v[h] = s;
  }
  if (h == 0) {
    float s = 0.f;
    for (int j = 0; j < H2; ++j) s += W_comb[j] * b_attn[j];
    *c = s;
  }
}

// ---------------- grid-wide barrier (persistent kernel) ----------------
static __device__ __forceinline__ void grid_sync(int* cnt, int* gen) {
  __threadfence();
  __syncthreads();
  if (threadIdx.x == 0) {
    int g = __hip_atomic_load(gen, __ATOMIC_RELAXED, __HIP_MEMORY_SCOPE_AGENT);
    if (atomicAdd(cnt, 1) == NBLK - 1) {
      __hip_atomic_store(cnt, 0, __ATOMIC_RELAXED, __HIP_MEMORY_SCOPE_AGENT);
      __hip_atomic_store(gen, g + 1, __ATOMIC_RELEASE, __HIP_MEMORY_SCOPE_AGENT);
    } else {
      while (__hip_atomic_load(gen, __ATOMIC_ACQUIRE, __HIP_MEMORY_SCOPE_AGENT) == g)
        __builtin_amdgcn_s_sleep(2);
    }
  }
  __syncthreads();
}

// ---------------- persistent fused GRU (both directions, all 200 steps) ----------------
// grid = 64 blocks (dir x 4 M-blocks x 8 N-blocks), block = 384 threads
// (12 waves = 4 M-strips x 3 gates). Per step: bf16 WMMA for x_t@W_ih^T (K=Epad)
// and h@W_hh^T (K=H, h tile async-staged in LDS), gate nonlinearities + masking,
// h update (f32 + bf16), out[t] write, then device-wide barrier.
__global__ __launch_bounds__(384)
void gru_persist_kernel(const unsigned short* __restrict__ inp_bf,
                        const unsigned short* __restrict__ Wih_f, const unsigned short* __restrict__ Whh_f,
                        const unsigned short* __restrict__ Wih_b, const unsigned short* __restrict__ Whh_b,
                        const float* __restrict__ bih_f, const float* __restrict__ bhh_f,
                        const float* __restrict__ bih_b, const float* __restrict__ bhh_b,
                        float* __restrict__ h32, unsigned short* __restrict__ hbf,
                        float* __restrict__ out_f, float* __restrict__ out_b,
                        const int* __restrict__ slen, int* __restrict__ bar) {
  __shared__ union {
    unsigned short hstage[64 * H];   // 64 KB: async-staged bf16 h tile (GEMM phase)
    float pre[4][64][64];            // 64 KB: r/z/xn/hn preactivations (epilogue)
  } smem;

  const int bx   = blockIdx.x;
  const int dir  = bx >> 5;
  const int mblk = bx & 3;
  const int nblk = (bx >> 2) & 7;
  const int tid  = threadIdx.x;
  const int wave = tid >> 5, lane = tid & 31;
  const int g    = wave % 3, ms = wave / 3;       // gate, M-strip
  const int rl   = lane & 15, half = lane >> 4;
  const int m0   = mblk * 64 + ms * 16;
  const int n0   = nblk * 64;

  const unsigned short* Wih = dir ? Wih_b : Wih_f;
  const unsigned short* Whh = dir ? Whh_b : Whh_f;
  const float* bih = dir ? bih_b : bih_f;
  const float* bhh = dir ? bhh_b : bhh_f;
  float*          h32d = h32 + (size_t)dir * B * H;
  unsigned short* hbfd = hbf + (size_t)dir * B * H;
  float*          op   = dir ? out_b : out_f;

  const int arow   = m0 + rl;        // A-fragment row (global batch index)
  const int arow_l = ms * 16 + rl;   // A-fragment row within staged LDS tile
  (void)arow;

  for (int s = 0; s < T; ++s) {
    const int t = dir ? (T - 1 - s) : s;
    const unsigned short* xt = inp_bf + (size_t)t * B * Epad;

    // Async-stage this block's 64-row slice of bf16 h into LDS (4096 x 16B).
    for (int c = tid; c < 64 * (H / 8); c += 384) {
      int row = c >> 6;
      int col = (c & 63) << 3;
      const unsigned short* gsrc = hbfd + ((size_t)(mblk * 64 + row)) * H + col;
      async_load_b128(gsrc, &smem.hstage[row * H + col]);
    }
    wait_async0();
    __syncthreads();

    v8f zero = {};
    v8f accX[4], accH[4];
#pragma unroll
    for (int j = 0; j < 4; ++j) { accX[j] = zero; accH[j] = zero; }

    // Phase 1: x_t @ W_ih[g]^T   (K = Epad = 320)
    for (int k0 = 0; k0 < Epad; k0 += 32) {
      const int kb = k0 + half * 8;
      v16bf a = load_frag(xt, Epad, arow, kb);
#pragma unroll
      for (int j = 0; j < 4; ++j) {
        v16bf bf = load_frag(Wih, Epad, g * H + n0 + j * 16 + rl, kb);
        accX[j] = __builtin_amdgcn_wmma_f32_16x16x32_bf16(false, a, false, bf,
                                                          (short)0, accX[j], false, false);
      }
    }
    // Phase 2: h @ W_hh[g]^T   (K = H = 512); A-operand from the LDS stage
    for (int k0 = 0; k0 < H; k0 += 32) {
      const int kb = k0 + half * 8;
      v16bf a = load_frag_ptr(&smem.hstage[arow_l * H + kb]);
#pragma unroll
      for (int j = 0; j < 4; ++j) {
        v16bf bf = load_frag(Whh, H, g * H + n0 + j * 16 + rl, kb);
        accH[j] = __builtin_amdgcn_wmma_f32_16x16x32_bf16(false, a, false, bf,
                                                          (short)0, accH[j], false, false);
      }
    }

    __syncthreads();   // stage buffer dead; safe to overwrite union with preactivations

    // Scatter preactivations (C/D layout: vgpr i -> m = i + 8*half, n = rl)
#pragma unroll
    for (int j = 0; j < 4; ++j) {
#pragma unroll
      for (int i = 0; i < 8; ++i) {
        int r = ms * 16 + i + half * 8;
        int c = j * 16 + rl;
        if (g == 2) { smem.pre[2][r][c] = accX[j][i]; smem.pre[3][r][c] = accH[j][i]; }
        else        { smem.pre[g][r][c] = accX[j][i] + accH[j][i]; }
      }
    }
    __syncthreads();

    // Gates + masking + state update
    for (int e = tid; e < 64 * 64; e += 384) {
      int ml = e >> 6, nl = e & 63;
      int m = mblk * 64 + ml;
      int n = n0 + nl;
      float pr = smem.pre[0][ml][nl] + bih[n]         + bhh[n];
      float pz = smem.pre[1][ml][nl] + bih[H + n]     + bhh[H + n];
      float xn = smem.pre[2][ml][nl] + bih[2 * H + n];
      float hn = smem.pre[3][ml][nl] + bhh[2 * H + n];
      float r  = 1.f / (1.f + __expf(-pr));
      float z  = 1.f / (1.f + __expf(-pz));
      float nn = tanhf(xn + r * hn);
      size_t hoff = (size_t)m * H + n;
      float hold = h32d[hoff];
      float hnew = (1.f - z) * nn + z * hold;
      bool  mv   = t < slen[m];
      float hkeep = mv ? hnew : hold;
      h32d[hoff] = hkeep;
      hbfd[hoff] = f2bf(hkeep);
      op[((size_t)t * B + m) * H + n] = mv ? hnew : 0.f;
    }

    grid_sync(bar, bar + 1);   // device-wide step barrier (replaces per-step launches)
  }
}

// ---------------- attention (folded) ----------------
__global__ __launch_bounds__(256)
void attn_kernel(const float* __restrict__ out_f, const float* __restrict__ out_b,
                 const float* __restrict__ v, const float* __restrict__ cptr,
                 const int* __restrict__ inv, const int* __restrict__ seq_len,
                 float* __restrict__ outp) {
  __shared__ float logits[T];
  __shared__ float aw[T];
  __shared__ float inv_sum_s;

  const int b = blockIdx.x;
  const int tid = threadIdx.x;
  const int wave = tid >> 5, lane = tid & 31;
  const int p = inv[b];
  const int L = seq_len[b];
  const float cbias = cptr[0];

  for (int t = wave; t < T; t += 8) {
    float s = 0.f;
    for (int h = lane; h < H2; h += 32) {
      float u = (h < H) ? out_f[((size_t)t * B + p) * H + h]
                        : out_b[((size_t)t * B + p) * H + (h - H)];
      s += u * v[h];
    }
#pragma unroll
    for (int off = 16; off > 0; off >>= 1) s += __shfl_down(s, off);
    if (lane == 0) logits[t] = s + cbias;
  }
  __syncthreads();

  if (tid == 0) {
    float mx = -1e30f;
    for (int t = 0; t < L; ++t) mx = fmaxf(mx, logits[t]);
    float sm = 0.f;
    for (int t = 0; t < T; ++t) {
      float e = (t < L) ? __expf(logits[t] - mx) : 0.f;
      aw[t] = e;
      sm += e;
    }
    inv_sum_s = 1.f / sm;
  }
  __syncthreads();

  const float isum = inv_sum_s;
  for (int h = tid; h < H2; h += 256) {
    float acc = 0.f;
    if (h < H) {
      for (int t = 0; t < L; ++t) acc += aw[t] * out_f[((size_t)t * B + p) * H + h];
    } else {
      int hh = h - H;
      for (int t = 0; t < L; ++t) acc += aw[t] * out_b[((size_t)t * B + p) * H + hh];
    }
    outp[(size_t)b * H2 + h] = acc * isum;   // d_out shape (B, 1, 2H)
  }
}

// ---------------- host launch ----------------
extern "C" void kernel_launch(void* const* d_in, const int* in_sizes, int n_in,
                              void* d_out, int out_size, void* d_ws, size_t ws_size,
                              hipStream_t stream) {
  const float* inp    = (const float*)d_in[0];
  const float* W_ih_f = (const float*)d_in[1];
  const float* W_hh_f = (const float*)d_in[2];
  const float* b_ih_f = (const float*)d_in[3];
  const float* b_hh_f = (const float*)d_in[4];
  const float* W_ih_b = (const float*)d_in[5];
  const float* W_hh_b = (const float*)d_in[6];
  const float* b_ih_b = (const float*)d_in[7];
  const float* b_hh_b = (const float*)d_in[8];
  const float* W_attn = (const float*)d_in[9];
  const float* b_attn = (const float*)d_in[10];
  const float* W_comb = (const float*)d_in[11];
  const int*   s_len  = (const int*)d_in[12];   // sorted_seq_length
  const int*   s_idx  = (const int*)d_in[13];   // sorted_idx
  const int*   q_len  = (const int*)d_in[14];   // seq_length_tensor
  float* outp = (float*)d_out;

  char* ws = (char*)d_ws;
  size_t off = 0;
  auto alloc = [&](size_t bytes) -> char* {
    char* p = ws + off;
    off = (off + bytes + 255) & ~(size_t)255;
    return p;
  };
  unsigned short* inp_bf = (unsigned short*)alloc((size_t)T * B * Epad * 2);
  unsigned short* wih_f  = (unsigned short*)alloc((size_t)G3H * Epad * 2);
  unsigned short* whh_f  = (unsigned short*)alloc((size_t)G3H * H * 2);
  unsigned short* wih_b  = (unsigned short*)alloc((size_t)G3H * Epad * 2);
  unsigned short* whh_b  = (unsigned short*)alloc((size_t)G3H * H * 2);
  float*          h32    = (float*)alloc((size_t)2 * B * H * 4);
  unsigned short* hbf    = (unsigned short*)alloc((size_t)2 * B * H * 2);
  float*          out_f  = (float*)alloc((size_t)T * B * H * 4);
  float*          out_b  = (float*)alloc((size_t)T * B * H * 4);
  float*          vvec   = (float*)alloc((size_t)H2 * 4);
  float*          cval   = (float*)alloc(4);
  int*            invp   = (int*)alloc((size_t)B * 4);
  int*            bar    = (int*)alloc(8);
  (void)ws_size; (void)in_sizes; (void)n_in; (void)out_size;

  // prep: bf16 conversions (padded), state init, inverse perm, folded attention vector
  {
    long n = (long)T * B * Epad;
    cvt_pad_kernel<<<(unsigned)((n + 255) / 256), 256, 0, stream>>>(inp, inp_bf, T * B, E, Epad);
    long nw = (long)G3H * Epad;
    cvt_pad_kernel<<<(unsigned)((nw + 255) / 256), 256, 0, stream>>>(W_ih_f, wih_f, G3H, E, Epad);
    cvt_pad_kernel<<<(unsigned)((nw + 255) / 256), 256, 0, stream>>>(W_ih_b, wih_b, G3H, E, Epad);
    long nh = (long)G3H * H;
    cvt_pad_kernel<<<(unsigned)((nh + 255) / 256), 256, 0, stream>>>(W_hh_f, whh_f, G3H, H, H);
    cvt_pad_kernel<<<(unsigned)((nh + 255) / 256), 256, 0, stream>>>(W_hh_b, whh_b, G3H, H, H);
    init_state_kernel<<<(2 * B * H + 255) / 256, 256, 0, stream>>>(h32, hbf, bar);
    inv_perm_kernel<<<1, B, 0, stream>>>(s_idx, invp);
    fold_attn_kernel<<<(H2 + 255) / 256, 256, 0, stream>>>(W_attn, b_attn, W_comb, vvec, cval);
  }

  // recurrence: ONE persistent launch, 200 steps with device-wide barriers
  gru_persist_kernel<<<NBLK, 384, 0, stream>>>(inp_bf, wih_f, whh_f, wih_b, whh_b,
                                               b_ih_f, b_hh_f, b_ih_b, b_hh_b,
                                               h32, hbf, out_f, out_b, s_len, bar);

  // folded attention + weighted sum
  attn_kernel<<<B, 256, 0, stream>>>(out_f, out_b, vvec, cval, invp, q_len, outp);
}
